// RNN_70480413327666
// MI455X (gfx1250) — compile-verified, matrix operand order checked by reading, not run on hardware
//
#include <hip/hip_runtime.h>

// CDNA5 / gfx1250: wave32, WMMA f32_16x16x32_f16.
typedef __attribute__((ext_vector_type(16))) _Float16 v16h;
typedef __attribute__((ext_vector_type(8)))  float    v8f;

#define T_STEPS 1024
#define BATCH   4096
#define HID     64
#define NCLS    5

static __device__ __forceinline__ float tanh_fast(float x) {
#if __has_builtin(__builtin_amdgcn_tanhf)
  return __builtin_amdgcn_tanhf(x);          // v_tanh_f32 (gfx1250 trans op)
#else
  float y;
  asm volatile("v_tanh_f32 %0, %1" : "=v"(y) : "v"(x));
  return y;
#endif
}

// K-index a lane's B/A element e maps to, per CDNA5 16-bit operand layout:
// e in [0,8):  k = half*8 + e ; e in [8,16): k = 16 + half*8 + (e-8)
static __device__ __forceinline__ int kmap(int e, int half) {
  return (e < 8) ? (half * 8 + e) : (16 + half * 8 + (e - 8));
}

__global__ __launch_bounds__(128, 1) void rnn_wmma_kernel(
    const float* __restrict__ x,      // [B, T]
    const float* __restrict__ w_ih,   // [H]
    const float* __restrict__ b_ih,   // [H]
    const float* __restrict__ w_hh,   // [H, H] row-major (j, k)
    const float* __restrict__ b_hh,   // [H]
    const float* __restrict__ w_head, // [C, H]
    const float* __restrict__ b_head, // [C]
    float* __restrict__ out)          // [B, C]
{
  const int lane = threadIdx.x & 31;
  const int wave = threadIdx.x >> 5;
  const int half = lane >> 4;       // which 16-lane half
  const int col  = lane & 15;
  const int wid  = blockIdx.x * 4 + wave;   // 256 waves total
  const int b    = wid * 16 + col;          // batch row this lane owns (dup across halves)

  // ---- A operands: w_hh tiles in f16, loop-invariant, live in VGPRs ----
  // A tile (jt, kc) is 16x32: row m = col -> j = jt*16+col, k = kc*32 + kmap(e,half)
  v16h a[4][2];
#pragma unroll
  for (int jt = 0; jt < 4; ++jt) {
    const int j = jt * 16 + col;
#pragma unroll
    for (int kc = 0; kc < 2; ++kc) {
      const float* wrow = w_hh + j * HID + kc * 32 + half * 8;
#pragma unroll
      for (int e = 0; e < 8; ++e) {
        a[jt][kc][e]     = (_Float16)wrow[e];        // k = kc*32 + half*8 + e
        a[jt][kc][e + 8] = (_Float16)wrow[16 + e];   // k = kc*32 + 16 + half*8 + e
      }
    }
  }

  // ---- per-lane xp coefficients (C/D layout: VGPR r -> j = jt*16 + half*8 + r) ----
  float wih_r[4][8], bias_r[4][8];
#pragma unroll
  for (int jt = 0; jt < 4; ++jt)
#pragma unroll
    for (int r = 0; r < 8; ++r) {
      const int j = jt * 16 + half * 8 + r;
      wih_r[jt][r]  = w_ih[j];
      bias_r[jt][r] = b_ih[j] + b_hh[j];
    }

  // ---- recurrent state as B operands (hi/lo compensated f16), h0 = 0 ----
  v16h bhi[2], blo[2];
#pragma unroll
  for (int kc = 0; kc < 2; ++kc)
#pragma unroll
    for (int e = 0; e < 16; ++e) { bhi[kc][e] = (_Float16)0.f; blo[kc][e] = (_Float16)0.f; }

  const float* xrow = x + (size_t)b * T_STEPS;   // 16B aligned (b*4096 bytes)
  float4 xv = *(const float4*)xrow;

  for (int t4 = 0; t4 < T_STEPS; t4 += 4) {
    float4 xnext = make_float4(0.f, 0.f, 0.f, 0.f);
    if (t4 + 4 < T_STEPS) xnext = *(const float4*)(xrow + t4 + 4);  // prefetch next group
    const float xs[4] = {xv.x, xv.y, xv.z, xv.w};

#pragma unroll
    for (int s = 0; s < 4; ++s) {
      const float xt = xs[s];

      // Two independent 2-deep WMMA chains per jt (instead of one 4-deep chain):
      //   acc_hi = 0  + A0*Bhi0 + A1*Bhi1     (C=0 -> inline-0 SRC2)
      //   acc_lo = xp + A0*Blo0 + A1*Blo1     (xp FMAs are off the critical path)
      //   pre-activation = acc_hi + acc_lo
      v8f acc_hi[4], acc_lo[4];
#pragma unroll
      for (int jt = 0; jt < 4; ++jt)
#pragma unroll
        for (int r = 0; r < 8; ++r) {
          acc_hi[jt][r] = 0.f;
          acc_lo[jt][r] = fmaf(xt, wih_r[jt][r], bias_r[jt][r]);
        }

#pragma unroll
      for (int jt = 0; jt < 4; ++jt) {
        acc_hi[jt] = __builtin_amdgcn_wmma_f32_16x16x32_f16(false, a[jt][0], false, bhi[0],
                                                            (short)0, acc_hi[jt], false, false);
        acc_hi[jt] = __builtin_amdgcn_wmma_f32_16x16x32_f16(false, a[jt][1], false, bhi[1],
                                                            (short)0, acc_hi[jt], false, false);
        acc_lo[jt] = __builtin_amdgcn_wmma_f32_16x16x32_f16(false, a[jt][0], false, blo[0],
                                                            (short)0, acc_lo[jt], false, false);
        acc_lo[jt] = __builtin_amdgcn_wmma_f32_16x16x32_f16(false, a[jt][1], false, blo[1],
                                                            (short)0, acc_lo[jt], false, false);
      }

      // tanh + rebuild B operands, purely lane-local:
      // D tile jt, row r holds j = jt*16 + half*8 + r, which is exactly
      // B element e = (jt&1)*8 + r of chunk kc = jt>>1 (k = kc*32 + kmap(e,half)).
#pragma unroll
      for (int jt = 0; jt < 4; ++jt)
#pragma unroll
        for (int r = 0; r < 8; ++r) {
          const float h  = tanh_fast(acc_hi[jt][r] + acc_lo[jt][r]);
          const _Float16 hh = (_Float16)h;
          const _Float16 hl = (_Float16)(h - (float)hh);
          const int kc = jt >> 1;
          const int e  = (jt & 1) * 8 + r;
          bhi[kc][e] = hh;
          blo[kc][e] = hl;
        }
    }
    xv = xnext;
  }

  // ---- head: out[b,c] = sum_j h[b,j] * w_head[c,j] + b_head[c] ----
  // Each lane holds 32 of the 64 j's (h = hi+lo); halves hold disjoint sets.
  float part[NCLS] = {0.f, 0.f, 0.f, 0.f, 0.f};
#pragma unroll
  for (int kc = 0; kc < 2; ++kc)
#pragma unroll
    for (int e = 0; e < 16; ++e) {
      const int j   = kc * 32 + kmap(e, half);
      const float h = (float)bhi[kc][e] + (float)blo[kc][e];
#pragma unroll
      for (int c = 0; c < NCLS; ++c)
        part[c] = fmaf(h, w_head[c * HID + j], part[c]);
    }
#pragma unroll
  for (int c = 0; c < NCLS; ++c) {
    const float tot = part[c] + __shfl_xor(part[c], 16, 32);
    if (half == 0) out[(size_t)b * NCLS + c] = tot + b_head[c];
  }
}

extern "C" void kernel_launch(void* const* d_in, const int* in_sizes, int n_in,
                              void* d_out, int out_size, void* d_ws, size_t ws_size,
                              hipStream_t stream) {
  const float* x      = (const float*)d_in[0];
  const float* w_ih   = (const float*)d_in[1];
  const float* b_ih   = (const float*)d_in[2];
  const float* w_hh   = (const float*)d_in[3];
  const float* b_hh   = (const float*)d_in[4];
  const float* w_head = (const float*)d_in[5];
  const float* b_head = (const float*)d_in[6];
  float* out = (float*)d_out;

  // 4096 batches / 16 per wave = 256 waves; 4 waves per 128-thread block -> 64 blocks.
  dim3 grid(BATCH / 16 / 4);
  dim3 block(128);
  rnn_wmma_kernel<<<grid, block, 0, stream>>>(x, w_ih, b_ih, w_hh, b_hh,
                                              w_head, b_head, out);
}
